// GaussLSTMModel_68083821576394
// MI455X (gfx1250) — matrix operand order repaired and checked by reading, not compiled, not run on hardware
//
#include <hip/hip_runtime.h>

// ---------------------------------------------------------------------------
// GP-LSTM LM for MI455X (gfx1250): f16 WMMA everywhere, f32 accumulate.
// ---------------------------------------------------------------------------

typedef __attribute__((ext_vector_type(16))) _Float16 v16h;
typedef __attribute__((ext_vector_type(8)))  _Float16 v8h;
typedef __attribute__((ext_vector_type(8)))  float    v8f;

#define T_DIM 256
#define B_DIM 8
#define V_DIM 50257
#define I_DIM 1024
#define H_DIM 1024

// ---- helpers --------------------------------------------------------------

__device__ __forceinline__ v8f wmma_f16(v16h a, v16h b, v8f c) {
    // D = A(16x32 f16) * B(32x16 f16) + C(16x16 f32)
    return __builtin_amdgcn_wmma_f32_16x16x32_f16(
        /*neg_a=*/false, a, /*neg_b=*/false, b,
        /*c_mod=*/(short)0, c, /*reuse_a=*/false, /*reuse_b=*/false);
}

// A-fragment (16x32, f16), contiguous k order (not AMD's table order): lane L holds row M=L%16,
// elements 0..15 -> K = k0 + 16*(L/16) + e  (contiguous, matches B)
// caller passes p = rowptr + k0 + 16*(L/16)
__device__ __forceinline__ v16h load_a_frag(const _Float16* __restrict__ p) {
    v8h lo = *(const v8h*)(p);
    v8h hi = *(const v8h*)(p + 8);
    return __builtin_shufflevector(lo, hi, 0,1,2,3,4,5,6,7,8,9,10,11,12,13,14,15);
}

__device__ __forceinline__ float sigm(float x) { return 1.0f / (1.0f + __expf(-x)); }

// ---- one-time conversion kernels ------------------------------------------

__global__ void cvt_f32_to_f16(const float* __restrict__ src,
                               _Float16* __restrict__ dst, size_t n) {
    size_t i = (size_t)blockIdx.x * blockDim.x + threadIdx.x;
    size_t stride = (size_t)gridDim.x * blockDim.x;
    for (; i < n; i += stride) dst[i] = (_Float16)src[i];
}

// embedding gather + f32->f16: e16[tb,:] = (f16) emb[x[tb],:]
__global__ void embed_kernel(const int* __restrict__ x,
                             const float* __restrict__ emb,
                             _Float16* __restrict__ e16) {
    int tb = blockIdx.x;                     // 0 .. T*B-1
    int row = x[tb];
    const float* src = emb + (size_t)row * I_DIM;
    _Float16* dst = e16 + (size_t)tb * I_DIM;
    for (int c = threadIdx.x; c < I_DIM; c += blockDim.x)
        dst[c] = (_Float16)src[c];
}

// ---- fused per-(layer,timestep) kernel ------------------------------------
// Grid: H/16 = 64 blocks, 256 threads (8 waves). Each block owns 16 columns
// of H. Each wave K-splits the 2048-deep reduction over 6 output tiles:
//   acc[0..3] : i,f,g,o gates  = x@wih[q]^T + h@whh[q]^T    (K=1024 each half)
//   acc[4]    : gc pre-act     = [x,h] @ wgc^T               (K=2048)
//   acc[5]    : go pre-act     = [x,h] @ wgo^T               (K=2048)
// Then LDS reduction across the 8 waves and the pointwise cell update done
// entirely in-register by wave 0 (lane = column, element r = batch row).

__global__ __launch_bounds__(256)
void step_kernel(const _Float16* __restrict__ x16,   // [8,1024] f16
                 const _Float16* __restrict__ h16,   // [8,1024] f16
                 const _Float16* __restrict__ wih,   // [4096,1024] f16
                 const _Float16* __restrict__ whh,   // [4096,1024] f16
                 const _Float16* __restrict__ wgc,   // [1024,2048] f16
                 const _Float16* __restrict__ wgo,   // [1024,2048] f16
                 const float* __restrict__ bih, const float* __restrict__ bhh,
                 const float* __restrict__ bgc, const float* __restrict__ bgo,
                 const float* __restrict__ blend,    // [2,2] row-major
                 float* __restrict__ c32,            // [8,1024] cell state (rw)
                 _Float16* __restrict__ yout,        // [8,1024] h output f16
                 float* __restrict__ hfin,           // [8,1024] final h (f32)
                 float* __restrict__ cfin,           // [8,1024] final c (f32)
                 int write_final) {
    const int lane = threadIdx.x & 31;
    const int wave = threadIdx.x >> 5;
    const int half = lane >> 4;       // K-group selector within wave
    const int lm   = lane & 15;       // A: row M ; B/D: column N
    const int jb   = blockIdx.x << 4; // column base in H

    // A rows 8..15 are padding: clamp to a valid row, rows 8..15 of D are
    // garbage but never stored. No divergence -> EXEC stays all-ones.
    const _Float16* xrow = x16 + (size_t)(lm & 7) * 1024;
    const _Float16* hrow = h16 + (size_t)(lm & 7) * 1024;

    v8f zero = {0.f,0.f,0.f,0.f,0.f,0.f,0.f,0.f};
    v8f acc[6] = {zero, zero, zero, zero, zero, zero};

    // K = 0..2047 ; first 1024 comes from x, second 1024 from h ("cat").
    for (int kc = wave * 32; kc < 2048; kc += 256) {
        const int ko = kc & 1023;
        const _Float16* asrc = (kc < 1024) ? xrow : hrow;
        v16h a = load_a_frag(asrc + ko + 16 * half);

        const _Float16* WG = (kc < 1024) ? wih : whh;   // gate weights
        const int boff = ko + 16 * half;
#pragma unroll
        for (int q = 0; q < 4; ++q) {
            v16h b = *(const v16h*)(WG + (size_t)(q * 1024 + jb + lm) * 1024 + boff);
            acc[q] = wmma_f16(a, b, acc[q]);
        }
        {   // GPNN sub-gates: full 2048-wide rows of wgc/wgo
            const size_t roff = (size_t)(jb + lm) * 2048 + kc + 16 * half;
            v16h bc = *(const v16h*)(wgc + roff);
            acc[4] = wmma_f16(a, bc, acc[4]);
            v16h bo = *(const v16h*)(wgo + roff);
            acc[5] = wmma_f16(a, bo, acc[5]);
        }
    }

    // cross-wave reduction of the K-split partial sums
    __shared__ float red[8][6][32][8];   // 48 KB
#pragma unroll
    for (int t = 0; t < 6; ++t)
#pragma unroll
        for (int r = 0; r < 8; ++r)
            red[wave][t][lane][r] = acc[t][r];
    __syncthreads();

    if (wave == 0) {
#pragma unroll
        for (int t = 0; t < 6; ++t)
#pragma unroll
            for (int r = 0; r < 8; ++r) {
                float s = 0.f;
#pragma unroll
                for (int w = 0; w < 8; ++w) s += red[w][t][lane][r];
                acc[t][r] = s;
            }

        if (lane < 16) {
            // lane = column n; element r = batch row (D layout, half 0 only)
            const int col = jb + lane;
            const float b_i = bih[col]        + bhh[col];
            const float b_f = bih[col + 1024] + bhh[col + 1024];
            const float b_g = bih[col + 2048] + bhh[col + 2048];
            const float b_o = bih[col + 3072] + bhh[col + 3072];
            const float bgcv = bgc[col], bgov = bgo[col];

            // softmax over blend rows
            const float e00 = __expf(blend[0]), e01 = __expf(blend[1]);
            const float e10 = __expf(blend[2]), e11 = __expf(blend[3]);
            const float p00 = e00 / (e00 + e01), p01 = e01 / (e00 + e01);
            const float p10 = e10 / (e10 + e11), p11 = e11 / (e10 + e11);

#pragma unroll
            for (int r = 0; r < B_DIM; ++r) {
                const float iv  = sigm(acc[0][r] + b_i);
                const float fv  = sigm(acc[1][r] + b_f);
                const float gv  = tanhf(acc[2][r] + b_g);
                const float ov  = sigm(acc[3][r] + b_o);
                const float gcv = tanhf(acc[4][r] + bgcv);
                const float gov = sigm(acc[5][r] + bgov);
                const float g = gv * p00 + gcv * p01;
                const float o = ov * p10 + gov * p11;
                const size_t idx = (size_t)r * 1024 + col;
                const float cnew = fv * c32[idx] + iv * g;
                const float hnew = o * tanhf(cnew);
                c32[idx]  = cnew;
                yout[idx] = (_Float16)hnew;
                if (write_final) { hfin[idx] = hnew; cfin[idx] = cnew; }
            }
        }
    }
}

// ---- decoder: out[2048, V] = Y[2048,1024] @ Wdec[V,1024]^T + b ------------
// Register-blocked: each wave owns a 64(M) x 64(N) region = 4x4 WMMA tiles.
// Per K-chunk: 8 fragment loads feed 16 WMMAs (~32 FLOP per cache byte).
// Block = 8 waves side by side in N (512 vocab cols); grid.x over vocab
// column groups, grid.y over 32 M groups (64 rows each).

__global__ __launch_bounds__(256)
void decode_kernel(const _Float16* __restrict__ Y,     // [2048,1024] f16
                   const _Float16* __restrict__ W,     // [V,1024] f16
                   const float* __restrict__ bias,     // [V]
                   float* __restrict__ out) {          // [2048, V]
    const int lane = threadIdx.x & 31;
    const int wave = threadIdx.x >> 5;
    const int half = lane >> 4;
    const int lm   = lane & 15;
    const int m0   = blockIdx.y << 6;                  // 64 rows / block
    const int c0   = blockIdx.x * 512 + wave * 64;     // 64 cols / wave

    const _Float16* arow[4];
#pragma unroll
    for (int mi = 0; mi < 4; ++mi)
        arow[mi] = Y + (size_t)(m0 + mi * 16 + lm) * 1024 + 16 * half;

    const _Float16* brow[4];
#pragma unroll
    for (int ni = 0; ni < 4; ++ni) {
        const int col  = c0 + ni * 16 + lm;
        const int wrow = (col < V_DIM) ? col : (V_DIM - 1);  // clamp: no divergence
        brow[ni] = W + (size_t)wrow * 1024 + 16 * half;
    }

    v8f zero = {0.f,0.f,0.f,0.f,0.f,0.f,0.f,0.f};
    v8f acc[4][4];
#pragma unroll
    for (int mi = 0; mi < 4; ++mi)
#pragma unroll
        for (int ni = 0; ni < 4; ++ni) acc[mi][ni] = zero;

    for (int k = 0; k < 1024; k += 32) {
        v16h a[4], b[4];
#pragma unroll
        for (int mi = 0; mi < 4; ++mi) a[mi] = load_a_frag(arow[mi] + k);
#pragma unroll
        for (int ni = 0; ni < 4; ++ni) b[ni] = *(const v16h*)(brow[ni] + k);
#pragma unroll
        for (int mi = 0; mi < 4; ++mi)
#pragma unroll
            for (int ni = 0; ni < 4; ++ni)
                acc[mi][ni] = wmma_f16(a[mi], b[ni], acc[mi][ni]);
    }

#pragma unroll
    for (int ni = 0; ni < 4; ++ni) {
        const int col = c0 + ni * 16 + lm;
        if (col < V_DIM) {
            const float bv = bias[col];
#pragma unroll
            for (int mi = 0; mi < 4; ++mi)
#pragma unroll
                for (int r = 0; r < 8; ++r)
                    out[(size_t)(m0 + mi * 16 + r + 8 * half) * V_DIM + col] =
                        acc[mi][ni][r] + bv;
        }
    }
}

// ---------------------------------------------------------------------------
// host launch
// ---------------------------------------------------------------------------

extern "C" void kernel_launch(void* const* d_in, const int* in_sizes, int n_in,
                              void* d_out, int out_size, void* d_ws, size_t ws_size,
                              hipStream_t stream) {
    (void)in_sizes; (void)n_in; (void)out_size; (void)ws_size;

    const int*   x    = (const int*)  d_in[0];
    const float* h0   = (const float*)d_in[1];
    const float* c0   = (const float*)d_in[2];
    const float* emb  = (const float*)d_in[3];
    const float* wdec = (const float*)d_in[4];
    const float* bdec = (const float*)d_in[5];
    const float* wihf[2]  = {(const float*)d_in[6],  (const float*)d_in[15]};
    const float* bih[2]   = {(const float*)d_in[7],  (const float*)d_in[16]};
    const float* whhf[2]  = {(const float*)d_in[8],  (const float*)d_in[17]};
    const float* bhh[2]   = {(const float*)d_in[9],  (const float*)d_in[18]};
    const float* blend[2] = {(const float*)d_in[10], (const float*)d_in[19]};
    const float* wgcf[2]  = {(const float*)d_in[11], (const float*)d_in[20]};
    const float* bgc[2]   = {(const float*)d_in[12], (const float*)d_in[21]};
    const float* wgof[2]  = {(const float*)d_in[13], (const float*)d_in[22]};
    const float* bgo[2]   = {(const float*)d_in[14], (const float*)d_in[23]};

    float* out_dec = (float*)d_out;
    float* out_h   = out_dec + (size_t)T_DIM * B_DIM * V_DIM;
    float* out_c   = out_h + 2 * B_DIM * H_DIM;

    // bump allocator over workspace (~163 MB total)
    char* ws = (char*)d_ws;
    size_t off = 0;
    auto alloc = [&](size_t bytes) -> void* {
        void* p = ws + off;
        off = (off + bytes + 255) & ~(size_t)255;
        return p;
    };

    const size_t BH = (size_t)B_DIM * H_DIM;                 // 8192
    _Float16* wdec16 = (_Float16*)alloc((size_t)V_DIM * H_DIM * 2);
    _Float16 *wih16[2], *whh16[2], *wgc16[2], *wgo16[2];
    for (int l = 0; l < 2; ++l) {
        wih16[l] = (_Float16*)alloc((size_t)4 * H_DIM * I_DIM * 2);
        whh16[l] = (_Float16*)alloc((size_t)4 * H_DIM * H_DIM * 2);
        wgc16[l] = (_Float16*)alloc((size_t)H_DIM * (I_DIM + H_DIM) * 2);
        wgo16[l] = (_Float16*)alloc((size_t)H_DIM * (I_DIM + H_DIM) * 2);
    }
    _Float16* e16   = (_Float16*)alloc((size_t)T_DIM * B_DIM * I_DIM * 2);
    _Float16* y0    = (_Float16*)alloc((size_t)T_DIM * B_DIM * H_DIM * 2);
    _Float16* y1    = (_Float16*)alloc((size_t)T_DIM * B_DIM * H_DIM * 2);
    _Float16* hin16 = (_Float16*)alloc(2 * BH * 2);
    float*    c32   = (float*)   alloc(2 * BH * 4);

    auto cvt = [&](const float* s, _Float16* d, size_t n) {
        int blocks = (int)((n + 2047) / 2048);
        if (blocks > 2048) blocks = 2048;
        cvt_f32_to_f16<<<blocks, 256, 0, stream>>>(s, d, n);
    };

    // one-time precision conversions (re-done every call; deterministic)
    cvt(wdec, wdec16, (size_t)V_DIM * H_DIM);
    for (int l = 0; l < 2; ++l) {
        cvt(wihf[l], wih16[l], (size_t)4 * H_DIM * I_DIM);
        cvt(whhf[l], whh16[l], (size_t)4 * H_DIM * H_DIM);
        cvt(wgcf[l], wgc16[l], (size_t)H_DIM * (I_DIM + H_DIM));
        cvt(wgof[l], wgo16[l], (size_t)H_DIM * (I_DIM + H_DIM));
    }
    cvt(h0, hin16, 2 * BH);

    embed_kernel<<<T_DIM * B_DIM, 256, 0, stream>>>(x, emb, e16);
    hipMemcpyAsync(c32, c0, 2 * BH * sizeof(float),
                   hipMemcpyDeviceToDevice, stream);

    // sequential recurrence: one fused kernel per (timestep, layer)
    for (int t = 0; t < T_DIM; ++t) {
        for (int l = 0; l < 2; ++l) {
            const _Float16* xin = (l == 0) ? (e16 + (size_t)t * BH)
                                           : (y0  + (size_t)t * BH);
            _Float16* yl = (l == 0) ? y0 : y1;
            const _Float16* hprev = (t == 0) ? (hin16 + l * BH)
                                             : (yl + (size_t)(t - 1) * BH);
            const int wf = (t == T_DIM - 1) ? 1 : 0;
            step_kernel<<<H_DIM / 16, 256, 0, stream>>>(
                xin, hprev, wih16[l], whh16[l], wgc16[l], wgo16[l],
                bih[l], bhh[l], bgc[l], bgo[l], blend[l],
                c32 + l * BH, yl + (size_t)t * BH,
                out_h + l * BH, out_c + l * BH, wf);
        }
    }

    // big decoder GEMM: 64x64 per wave, 512 cols x 64 rows per block
    dim3 dgrid((V_DIM + 511) / 512, (T_DIM * B_DIM) / 64);
    decode_kernel<<<dgrid, 256, 0, stream>>>(y1, wdec16, bdec, out_dec);
}
